// TargetAwareRefinementBlock_81784767250888
// MI455X (gfx1250) — matrix operand order repaired and checked by reading, not compiled, not run on hardware
//
#include <hip/hip_runtime.h>
#include <hip/hip_bf16.h>
#include <math.h>

// ---------------------------------------------------------------------------
// TargetAwareRefinementBlock for MI455X (gfx1250), bf16 WMMA implementation.
// B=8 T=2048 H=1024 P=64 E=3 RH=2048, all GEMMs via v_wmma_f32_16x16x32_bf16.
// Round 2: async global->LDS staging (global_load_async_to_lds_b128 /
// s_wait_asynccnt) + LDS double buffering + software-pipelined B staging.
// ---------------------------------------------------------------------------

typedef __attribute__((ext_vector_type(16))) __bf16 v16bf;
typedef __attribute__((ext_vector_type(8)))  float  v8f;

#define TM 128
#define TN 128
#define TK 32
#define LDPAD 8                    // pad keeps row stride 16B-aligned (80 bytes)
#define LSTR (TK + LDPAD)

__device__ __forceinline__ float  bf2f(__bf16 h) { return (float)h; }
__device__ __forceinline__ __bf16 f2bf(float f)  { return (__bf16)f; }

// gfx1250 async copy: GLOBAL memory -> LDS, 16B per lane, tracked by ASYNCcnt.
__device__ __forceinline__ void async_copy_b128(unsigned lds_addr, const void* gaddr)
{
    asm volatile("global_load_async_to_lds_b128 %0, %1, off"
                 :: "v"(lds_addr), "v"(gaddr) : "memory");
}
__device__ __forceinline__ void wait_asynccnt0()
{
    asm volatile("s_wait_asynccnt 0" ::: "memory");
}
// flat shared pointer -> wave-relative LDS byte offset (low 32 bits of the flat address)
__device__ __forceinline__ unsigned lds_off(const void* p)
{
    return (unsigned)(unsigned long long)p;
}

// ---------------------------------------------------------------------------
// Generic batched GEMM: C[z] = act(alpha * A[z] @ B[z] + bias[z]) * rowScale
//   A: [M,K] bf16 row-major (lda), B: [K,N] bf16 row-major (ldb)
//   Cf: optional f32 out (store or atomic-add), Ch: optional bf16 out
// Assumes M % 128 == 0 and K % 32 == 0 (true for every call site here);
// N is guarded (needed for the N=64 attention GEMMs).
// ---------------------------------------------------------------------------
__global__ __launch_bounds__(256)
void gemm_bf16_wmma(const __bf16* __restrict__ A, long sAz, int lda,
                    const __bf16* __restrict__ Bm, long sBz, int ldb,
                    float* __restrict__ Cf, long sCfz, int ldcf,
                    __bf16* __restrict__ Ch, long sChz, int ldch,
                    const float* __restrict__ bias, int sBiasZ,
                    const float* __restrict__ rowScale, int rsStride,
                    int M, int N, int K, float alpha, int act, int atomicAcc)
{
    // double-buffered tiles: 2 * (128*40 + 128*40) * 2B = 40 KB
    __shared__ __align__(16) __bf16 lA[2][TM][LSTR];
    __shared__ __align__(16) __bf16 lB[2][TN][LSTR];   // transposed: lB[..][n][k]

    const int tid  = threadIdx.x;
    const int lane = tid & 31;
    const int w    = tid >> 5;        // wave 0..7
    const int wm   = w & 3;           // 4 waves tile M
    const int wn   = w >> 2;          // 2 waves tile N
    const int z    = blockIdx.z;
    const int mbase = blockIdx.y * TM;
    const int nbase = blockIdx.x * TN;

    const __bf16* Ab = A  + (size_t)z * sAz;
    const __bf16* Bb = Bm + (size_t)z * sBz;

    const v8f vzero = {0.f, 0.f, 0.f, 0.f, 0.f, 0.f, 0.f, 0.f};
    v8f acc[2][4];
#pragma unroll
    for (int i = 0; i < 2; ++i)
#pragma unroll
        for (int j = 0; j < 4; ++j) acc[i][j] = vzero;

    const int lrow  = lane & 15;            // m/n within 16x16 tile
    const int khalf = (lane >> 4) * 8;      // A-fragment K offset per half-wave
    const int kbB   = (lane >> 4) * 16;     // unused: B fragments also use khalf

    // per-thread staging coordinates (2 chunks each for A and B)
    const int aR[2]  = { (tid + 0) >> 2,        (tid + 256) >> 2 };
    const int aC[2]  = { ((tid + 0) & 3) * 8,   ((tid + 256) & 3) * 8 };
    const int bK[2]  = { (tid + 0) >> 4,        (tid + 256) >> 4 };
    const int bN[2]  = { ((tid + 0) & 15) * 8,  ((tid + 256) & 15) * 8 };

    // ---- helpers -----------------------------------------------------------
    auto stageA_async = [&](int buf, int k0) {
#pragma unroll
        for (int i = 0; i < 2; ++i) {
            unsigned la = lds_off(&lA[buf][aR[i]][aC[i]]);
            const void* ga = (const void*)(Ab + (size_t)(mbase + aR[i]) * lda + k0 + aC[i]);
            async_copy_b128(la, ga);
        }
    };
    auto loadB_regs = [&](int k0, uint4* bv) {
#pragma unroll
        for (int i = 0; i < 2; ++i) {
            uint4 val = make_uint4(0u, 0u, 0u, 0u);
            if (nbase + bN[i] < N)
                val = *(const uint4*)(Bb + (size_t)(k0 + bK[i]) * ldb + nbase + bN[i]);
            bv[i] = val;
        }
    };
    auto scatterB = [&](int buf, const uint4* bv) {
#pragma unroll
        for (int i = 0; i < 2; ++i) {
            __bf16 tmp[8];
            *(uint4*)tmp = bv[i];
#pragma unroll
            for (int j = 0; j < 8; ++j) lB[buf][bN[i] + j][bK[i]] = tmp[j];
        }
    };
    auto compute = [&](int buf) {
        union Frag { uint4 u[2]; v16bf v; };
        Frag fa[2], fb[4];
#pragma unroll
        for (int im = 0; im < 2; ++im) {
            int m = wm * 32 + im * 16 + lrow;
            fa[im].u[0] = *(const uint4*)&lA[buf][m][khalf];       // K = khalf..+7
            fa[im].u[1] = *(const uint4*)&lA[buf][m][khalf + 16];  // K = khalf+16..+23
        }
#pragma unroll
        for (int in = 0; in < 4; ++in) {
            int n = wn * 64 + in * 16 + lrow;
            fb[in].u[0] = *(const uint4*)&lB[buf][n][khalf];       // K = khalf..+7
            fb[in].u[1] = *(const uint4*)&lB[buf][n][khalf + 16];  // K = khalf+16..+23
        }
#pragma unroll
        for (int im = 0; im < 2; ++im)
#pragma unroll
            for (int in = 0; in < 4; ++in)
                acc[im][in] = __builtin_amdgcn_wmma_f32_16x16x32_bf16(
                    false, fa[im].v, false, fb[in].v,
                    (short)0, acc[im][in], false, false);
    };

    // ---- software-pipelined main loop --------------------------------------
    const int kIter = K / TK;
    {
        // prologue: fill buffer 0 with tile 0
        stageA_async(0, 0);
        uint4 bv[2];
        loadB_regs(0, bv);
        scatterB(0, bv);
        wait_asynccnt0();
        __syncthreads();
    }
    int buf = 0;
    for (int it = 0; it < kIter; ++it) {
        const bool hasNext = (it + 1 < kIter);
        uint4 nbv[2];
        if (hasNext) {
            stageA_async(buf ^ 1, (it + 1) * TK);   // ASYNCcnt traffic
            loadB_regs((it + 1) * TK, nbv);         // LOADcnt traffic, consumed late
        }
        compute(buf);                               // ds_load_b128 + 8x WMMA
        if (hasNext) {
            scatterB(buf ^ 1, nbv);
            wait_asynccnt0();
        }
        __syncthreads();
        buf ^= 1;
    }

    // ---- epilogue: C/D layout: VGPR r -> row (lane<16 ? r : r+8), col lane%16
    const int rowoff = (lane >> 4) * 8;
#pragma unroll
    for (int im = 0; im < 2; ++im) {
#pragma unroll
        for (int in = 0; in < 4; ++in) {
            float* av = (float*)&acc[im][in];
#pragma unroll
            for (int r = 0; r < 8; ++r) {
                int row = mbase + wm * 32 + im * 16 + rowoff + r;
                int col = nbase + wn * 64 + in * 16 + lrow;
                if (col >= N) continue;
                float val = av[r] * alpha;
                if (bias)     val += bias[(size_t)z * sBiasZ + col];
                if (act == 1) val = val / (1.0f + __expf(-val));       // SiLU
                if (rowScale) val *= rowScale[(size_t)row * rsStride + z];
                if (Cf) {
                    float* dst = Cf + (size_t)z * sCfz + (size_t)row * ldcf + col;
                    if (atomicAcc) atomicAdd(dst, val);
                    else           *dst = val;
                }
                if (Ch)
                    Ch[(size_t)z * sChz + (size_t)row * ldch + col] = f2bf(val);
            }
        }
    }
}

// ---------------------------------------------------------------------------
// LayerNorm over H=1024; writes bf16 result directly into combined[:, 0:1024].
// ---------------------------------------------------------------------------
__global__ __launch_bounds__(256)
void ln_pack_kernel(const float* __restrict__ x, const float* __restrict__ gamma,
                    const float* __restrict__ beta, __bf16* __restrict__ combined)
{
    const long row = blockIdx.x;
    const float* xr = x + row * 1024;
    __shared__ float red[256];
    float v[4];
    float s = 0.f;
#pragma unroll
    for (int i = 0; i < 4; ++i) { v[i] = xr[threadIdx.x + 256 * i]; s += v[i]; }
    red[threadIdx.x] = s; __syncthreads();
    for (int off = 128; off > 0; off >>= 1) {
        if (threadIdx.x < off) red[threadIdx.x] += red[threadIdx.x + off];
        __syncthreads();
    }
    float mu = red[0] * (1.0f / 1024.0f);
    __syncthreads();
    s = 0.f;
#pragma unroll
    for (int i = 0; i < 4; ++i) { float d = v[i] - mu; s += d * d; }
    red[threadIdx.x] = s; __syncthreads();
    for (int off = 128; off > 0; off >>= 1) {
        if (threadIdx.x < off) red[threadIdx.x] += red[threadIdx.x + off];
        __syncthreads();
    }
    float rs = rsqrtf(red[0] * (1.0f / 1024.0f) + 1e-5f);
    __bf16* outr = combined + row * 3072;
#pragma unroll
    for (int i = 0; i < 4; ++i) {
        int h = threadIdx.x + 256 * i;
        outr[h] = f2bf((v[i] - mu) * rs * gamma[h] + beta[h]);
    }
}

// anchor broadcast into combined[:, 2048:3072]
__global__ void anchor_pack_kernel(const float* __restrict__ anchor,
                                   __bf16* __restrict__ combined, long n)
{
    for (long idx = blockIdx.x * 256L + threadIdx.x; idx < n; idx += gridDim.x * 256L) {
        long row = idx >> 10;            // / H
        int  h   = (int)(idx & 1023);
        long b   = row >> 11;            // / T
        combined[row * 3072 + 2048 + h] = f2bf(anchor[b * 1024 + h]);
    }
}

__global__ void cvt_f32_bf16_kernel(const float* __restrict__ in,
                                    __bf16* __restrict__ out, long n)
{
    for (long i = blockIdx.x * 256L + threadIdx.x; i < n; i += gridDim.x * 256L)
        out[i] = f2bf(in[i]);
}

// kT[b][h][p] = k[b][p][h]   (for the scores GEMM B operand)
__global__ void ktrans_kernel(const __bf16* __restrict__ kin,
                              __bf16* __restrict__ kT, long n)
{
    for (long i = blockIdx.x * 256L + threadIdx.x; i < n; i += gridDim.x * 256L) {
        long b = i / (64L * 1024L);
        long r = i - b * 64L * 1024L;
        int  p = (int)(r >> 10);
        int  h = (int)(r & 1023);
        kT[b * 1024L * 64L + (long)h * 64 + p] = kin[i];
    }
}

// softmax over P=64, one block (64 threads) per token row; emits bf16 weights
__global__ __launch_bounds__(64)
void softmax_p_kernel(const float* __restrict__ scores, __bf16* __restrict__ wts)
{
    const long row = blockIdx.x;
    __shared__ float red[64];
    float v = scores[row * 64 + threadIdx.x];
    red[threadIdx.x] = v; __syncthreads();
    for (int off = 32; off > 0; off >>= 1) {
        if (threadIdx.x < off) red[threadIdx.x] = fmaxf(red[threadIdx.x], red[threadIdx.x + off]);
        __syncthreads();
    }
    float m = red[0]; __syncthreads();
    float e = __expf(v - m);
    red[threadIdx.x] = e; __syncthreads();
    for (int off = 32; off > 0; off >>= 1) {
        if (threadIdx.x < off) red[threadIdx.x] += red[threadIdx.x + off];
        __syncthreads();
    }
    wts[row * 64 + threadIdx.x] = f2bf(e / red[0]);
}

// router: logits = combined @ W[3072,3] + b, softmax over E=3
__global__ __launch_bounds__(128)
void router_kernel(const __bf16* __restrict__ combined,
                   const float* __restrict__ W, const float* __restrict__ b,
                   float* __restrict__ probs)
{
    const long row = blockIdx.x;
    const __bf16* c = combined + row * 3072;
    float a0 = 0.f, a1 = 0.f, a2 = 0.f;
    for (int i = threadIdx.x; i < 3072; i += 128) {
        float cv = bf2f(c[i]);
        a0 += cv * W[i * 3 + 0];
        a1 += cv * W[i * 3 + 1];
        a2 += cv * W[i * 3 + 2];
    }
    __shared__ float r0[128], r1[128], r2[128];
    r0[threadIdx.x] = a0; r1[threadIdx.x] = a1; r2[threadIdx.x] = a2;
    __syncthreads();
    for (int off = 64; off > 0; off >>= 1) {
        if (threadIdx.x < off) {
            r0[threadIdx.x] += r0[threadIdx.x + off];
            r1[threadIdx.x] += r1[threadIdx.x + off];
            r2[threadIdx.x] += r2[threadIdx.x + off];
        }
        __syncthreads();
    }
    if (threadIdx.x == 0) {
        float l0 = r0[0] + b[0], l1 = r1[0] + b[1], l2 = r2[0] + b[2];
        float m  = fmaxf(l0, fmaxf(l1, l2));
        float e0 = __expf(l0 - m), e1 = __expf(l1 - m), e2 = __expf(l2 - m);
        float inv = 1.0f / (e0 + e1 + e2);
        probs[row * 3 + 0] = e0 * inv;
        probs[row * 3 + 1] = e1 * inv;
        probs[row * 3 + 2] = e2 * inv;
    }
}

// routed[row,h] = sum_e probs[row,e] * e_b2[e,h]   (init before atomic GEMM adds)
__global__ void routed_init_kernel(const float* __restrict__ probs,
                                   const float* __restrict__ b2,
                                   float* __restrict__ routed, long n)
{
    for (long i = blockIdx.x * 256L + threadIdx.x; i < n; i += gridDim.x * 256L) {
        long row = i >> 10;
        int  h   = (int)(i & 1023);
        routed[i] = probs[row * 3 + 0] * b2[h]
                  + probs[row * 3 + 1] * b2[1024 + h]
                  + probs[row * 3 + 2] * b2[2048 + h];
    }
}

// out = x + sigmoid(glin) * proj
__global__ void final_kernel(const float* __restrict__ x,
                             const float* __restrict__ glin,
                             const float* __restrict__ proj,
                             float* __restrict__ out, long n)
{
    for (long i = blockIdx.x * 256L + threadIdx.x; i < n; i += gridDim.x * 256L) {
        float g = 1.0f / (1.0f + __expf(-glin[i]));
        out[i] = x[i] + g * proj[i];
    }
}

// ---------------------------------------------------------------------------
extern "C" void kernel_launch(void* const* d_in, const int* in_sizes, int n_in,
                              void* d_out, int out_size, void* d_ws, size_t ws_size,
                              hipStream_t stream)
{
    const int  B = 8, T = 2048, H = 1024, P = 64, E = 3, RH = 2048;
    const long BT = (long)B * T;
    const int  C3 = 3 * H;

    const float* x      = (const float*)d_in[0];
    const float* anchor = (const float*)d_in[1];
    const float* proto  = (const float*)d_in[2];
    const float* lng    = (const float*)d_in[3];
    const float* lnb    = (const float*)d_in[4];
    const float* Wq     = (const float*)d_in[5];
    const float* Wk     = (const float*)d_in[6];
    const float* Wv     = (const float*)d_in[7];
    const float* rW     = (const float*)d_in[8];
    const float* rb     = (const float*)d_in[9];
    const float* ew1    = (const float*)d_in[10];
    const float* eb1    = (const float*)d_in[11];
    const float* ew2    = (const float*)d_in[12];
    const float* eb2    = (const float*)d_in[13];
    const float* gw1    = (const float*)d_in[14];
    const float* gb1    = (const float*)d_in[15];
    const float* gw2    = (const float*)d_in[16];
    const float* gb2    = (const float*)d_in[17];
    const float* ow     = (const float*)d_in[18];
    const float* ob     = (const float*)d_in[19];
    float* out = (float*)d_out;

    // bump allocator over d_ws
    char*  ws  = (char*)d_ws;
    size_t off = 0;
    auto alloc = [&](size_t bytes) -> char* {
        char* p = ws + off;
        off += (bytes + 255) & ~(size_t)255;
        return p;
    };
    __bf16* combined  = (__bf16*)alloc(BT * C3 * 2);            // [BT,3072] bf16
    __bf16* q_bf      = (__bf16*)alloc(BT * H * 2);
    __bf16* proto_bf  = (__bf16*)alloc((long)B * P * H * 2);
    __bf16* wq_bf     = (__bf16*)alloc((long)H * H * 2);
    __bf16* wk_bf     = (__bf16*)alloc((long)H * H * 2);
    __bf16* wv_bf     = (__bf16*)alloc((long)H * H * 2);
    __bf16* k_bf      = (__bf16*)alloc((long)B * P * H * 2);
    __bf16* v_bf      = (__bf16*)alloc((long)B * P * H * 2);
    __bf16* kT_bf     = (__bf16*)alloc((long)B * H * P * 2);
    float*  scores    = (float*)alloc(BT * P * 4);
    __bf16* wts_bf    = (__bf16*)alloc(BT * P * 2);
    float*  probs     = (float*)alloc(BT * E * 4);
    __bf16* ew1_bf    = (__bf16*)alloc((long)E * C3 * RH * 2);
    __bf16* ew2_bf    = (__bf16*)alloc((long)E * RH * H * 2);
    __bf16* gw1_bf    = (__bf16*)alloc((long)C3 * H * 2);
    __bf16* gw2_bf    = (__bf16*)alloc((long)H * H * 2);
    __bf16* outw_bf   = (__bf16*)alloc((long)H * H * 2);
    __bf16* h_bf      = (__bf16*)alloc((long)E * BT * RH * 2);  // biggest buffer
    float*  routed    = (float*)alloc(BT * H * 4);
    __bf16* routed_bf = (__bf16*)alloc(BT * H * 2);
    __bf16* gh_bf     = (__bf16*)alloc(BT * H * 2);
    float*  glin      = (float*)alloc(BT * H * 4);
    float*  proj      = (float*)alloc(BT * H * 4);
    (void)ws_size; (void)in_sizes; (void)n_in; (void)out_size;

    auto gemm = [&](const __bf16* A, long sAz, int lda,
                    const __bf16* Bm, long sBz, int ldb,
                    float* Cf, long sCfz, int ldcf,
                    __bf16* Ch, long sChz, int ldch,
                    const float* bias, int sBiasZ,
                    const float* rowScale, int rsStride,
                    int M, int N, int K, int Z,
                    float alpha, int act, int atomicAcc) {
        dim3 grid((N + TN - 1) / TN, (M + TM - 1) / TM, Z);
        gemm_bf16_wmma<<<grid, 256, 0, stream>>>(
            A, sAz, lda, Bm, sBz, ldb, Cf, sCfz, ldcf, Ch, sChz, ldch,
            bias, sBiasZ, rowScale, rsStride, M, N, K, alpha, act, atomicAcc);
    };

    // --- weight / activation converts to bf16 --------------------------------
    cvt_f32_bf16_kernel<<<1024, 256, 0, stream>>>(Wq,  wq_bf,   (long)H * H);
    cvt_f32_bf16_kernel<<<1024, 256, 0, stream>>>(Wk,  wk_bf,   (long)H * H);
    cvt_f32_bf16_kernel<<<1024, 256, 0, stream>>>(Wv,  wv_bf,   (long)H * H);
    cvt_f32_bf16_kernel<<<1024, 256, 0, stream>>>(proto, proto_bf, (long)B * P * H);
    cvt_f32_bf16_kernel<<<4096, 256, 0, stream>>>(ew1, ew1_bf,  (long)E * C3 * RH);
    cvt_f32_bf16_kernel<<<2048, 256, 0, stream>>>(ew2, ew2_bf,  (long)E * RH * H);
    cvt_f32_bf16_kernel<<<1024, 256, 0, stream>>>(gw1, gw1_bf,  (long)C3 * H);
    cvt_f32_bf16_kernel<<<1024, 256, 0, stream>>>(gw2, gw2_bf,  (long)H * H);
    cvt_f32_bf16_kernel<<<1024, 256, 0, stream>>>(ow,  outw_bf, (long)H * H);

    // --- LayerNorm + concat slices ------------------------------------------
    ln_pack_kernel<<<(unsigned)BT, 256, 0, stream>>>(x, lng, lnb, combined);
    anchor_pack_kernel<<<4096, 256, 0, stream>>>(anchor, combined, BT * H);

    // --- retriever -----------------------------------------------------------
    // q = nx @ Wq : A = combined slice 0 (lda=3072)
    gemm(combined, 0, C3, wq_bf, 0, H,
         nullptr, 0, 0, q_bf, 0, H,
         nullptr, 0, nullptr, 0, (int)BT, H, H, 1, 1.0f, 0, 0);
    // k/v = proto @ Wk/Wv : M = B*P = 512
    gemm(proto_bf, 0, H, wk_bf, 0, H,
         nullptr, 0, 0, k_bf, 0, H,
         nullptr, 0, nullptr, 0, B * P, H, H, 1, 1.0f, 0, 0);
    gemm(proto_bf, 0, H, wv_bf, 0, H,
         nullptr, 0, 0, v_bf, 0, H,
         nullptr, 0, nullptr, 0, B * P, H, H, 1, 1.0f, 0, 0);
    ktrans_kernel<<<512, 256, 0, stream>>>(k_bf, kT_bf, (long)B * P * H);
    // scores[b] = q[b] @ kT[b] * H^-0.5 : Z=B, M=T, N=P, K=H
    gemm(q_bf, (long)T * H, H, kT_bf, (long)H * P, P,
         scores, (long)T * P, P, nullptr, 0, 0,
         nullptr, 0, nullptr, 0, T, P, H, B, 0.03125f, 0, 0);
    softmax_p_kernel<<<(unsigned)BT, 64, 0, stream>>>(scores, wts_bf);
    // evidence[b] = wts[b] @ v[b], written into combined[:, 1024:2048]
    gemm(wts_bf, (long)T * P, P, v_bf, (long)P * H, H,
         nullptr, 0, 0, combined + H, (long)T * C3, C3,
         nullptr, 0, nullptr, 0, T, H, P, B, 1.0f, 0, 0);

    // --- router --------------------------------------------------------------
    router_kernel<<<(unsigned)BT, 128, 0, stream>>>(combined, rW, rb, probs);
    routed_init_kernel<<<4096, 256, 0, stream>>>(probs, eb2, routed, BT * H);

    // --- experts (the compute-dominant GEMMs) --------------------------------
    // h[e] = silu(combined @ e_w1[e] + e_b1[e])  : Z=E, shared A (sAz=0)
    gemm(combined, 0, C3, ew1_bf, (long)C3 * RH, RH,
         nullptr, 0, 0, h_bf, BT * (long)RH, RH,
         eb1, RH, nullptr, 0, (int)BT, RH, C3, E, 1.0f, 1, 0);
    // routed += probs[:,e] * (h[e] @ e_w2[e])  : atomic f32 accumulate
    gemm(h_bf, BT * (long)RH, RH, ew2_bf, (long)RH * H, H,
         routed, 0, H, nullptr, 0, 0,
         nullptr, 0, probs, E, (int)BT, H, RH, E, 1.0f, 0, 1);

    // --- gate MLP ------------------------------------------------------------
    gemm(combined, 0, C3, gw1_bf, 0, H,
         nullptr, 0, 0, gh_bf, 0, H,
         gb1, 0, nullptr, 0, (int)BT, H, C3, 1, 1.0f, 1, 0);
    gemm(gh_bf, 0, H, gw2_bf, 0, H,
         glin, 0, H, nullptr, 0, 0,
         gb2, 0, nullptr, 0, (int)BT, H, H, 1, 1.0f, 0, 0);

    // --- out projection + gated residual ------------------------------------
    cvt_f32_bf16_kernel<<<4096, 256, 0, stream>>>(routed, routed_bf, BT * H);
    gemm(routed_bf, 0, H, outw_bf, 0, H,
         proj, 0, H, nullptr, 0, 0,
         ob, 0, nullptr, 0, (int)BT, H, H, 1, 1.0f, 0, 0);
    final_kernel<<<4096, 256, 0, stream>>>(x, glin, proj, out, BT * H);
}
